// GNN_GumbelSelector_24567212933208
// MI455X (gfx1250) — compile-verified, matrix operand order checked by reading, not run on hardware
//
#include <hip/hip_runtime.h>
#include <hip/hip_bf16.h>

typedef __attribute__((ext_vector_type(16))) __bf16 v16bf;
typedef __attribute__((ext_vector_type(8)))  float  v8f;

#define NN 50000
#define NE 800000
#define FEAT 128
#define HID 256
#define NG 64

// ---------- helpers ----------
__device__ __forceinline__ unsigned short f2bf(float f) {
    unsigned int u = __float_as_uint(f);
    unsigned int r = u + 0x7FFFu + ((u >> 16) & 1u);   // round-to-nearest-even
    return (unsigned short)(r >> 16);
}

// ---------- trivial kernels ----------
__global__ void k_zero(float* p, int n) {
    int i = blockIdx.x * blockDim.x + threadIdx.x;
    if (i < n) p[i] = 0.0f;
}

__global__ void k_deg(const int* dst, float* deg, int e) {
    int i = blockIdx.x * blockDim.x + threadIdx.x;
    if (i < e) atomicAdd(&deg[dst[i]], 1.0f);
}

__global__ void k_node_norms(float* dinv, float* nself, int n) {
    int i = blockIdx.x * blockDim.x + threadIdx.x;
    if (i < n) {
        float r = rsqrtf(dinv[i] + 1.0f);  // deg(+self loop)
        dinv[i] = r;
        nself[i] = r * r;
    }
}

__global__ void k_edge_norms(const int* src, const int* dst, const float* dinv,
                             float* ne, int e) {
    int i = blockIdx.x * blockDim.x + threadIdx.x;
    if (i < e) ne[i] = dinv[src[i]] * dinv[dst[i]];
}

// ---------- WMMA GEMM: C[N,F] = A[N,K] @ W[K,F] (+bias)(+relu), bf16 compute / f32 acc ----------
#define ASTR 36   // ushort stride per LDS row (72B, dword aligned, conflict-padded)

__global__ __launch_bounds__(256)
void k_gemm_wmma(const float* __restrict__ A, const float* __restrict__ W,
                 const float* __restrict__ bias, float* __restrict__ C,
                 int N, int K, int F, int relu) {
    __shared__ unsigned short As[64 * ASTR];  // A tile: 64 rows x 32 k (bf16)
    __shared__ unsigned short Bs[64 * ASTR];  // W tile transposed: 64 cols x 32 k (bf16)

    const int mBase = blockIdx.y * 64;
    const int nBase = blockIdx.x * 64;
    const int tid   = threadIdx.x;
    const int lane  = tid & 31;
    const int wid   = tid >> 5;
    const int waveM = wid & 3;     // 4 wave-rows  -> 64 rows
    const int waveN = wid >> 2;    // 2 wave-cols  -> 64 cols (2x 16x16 per wave)
    const int half  = lane >> 4;   // CDNA5 16-bit operand striping
    const int l15   = lane & 15;

    v8f acc0 = {}, acc1 = {};

    for (int k0 = 0; k0 < K; k0 += 32) {
        // stage A tile (f32 -> bf16), coalesced
        for (int i = tid; i < 64 * 32; i += 256) {
            int r = i >> 5, kk = i & 31;
            int gr = mBase + r;
            float v = (gr < N) ? A[(long)gr * K + (k0 + kk)] : 0.0f;
            As[r * ASTR + kk] = f2bf(v);
        }
        // stage W tile transposed: Bs[col][k]
        for (int i = tid; i < 64 * 32; i += 256) {
            int kk = i >> 6, c = i & 63;
            Bs[c * ASTR + kk] = f2bf(W[(long)(k0 + kk) * F + (nBase + c)]);
        }
        __syncthreads();

        union Frag { v16bf v; unsigned int u[8]; } af, bf0, bf1;
        // A fragment: row = waveM*16+l15 ; K set {kb..kb+7} U {16+kb..16+kb+7}, kb = half*8
        const unsigned int* arow  = (const unsigned int*)&As[(waveM * 16 + l15) * ASTR];
        const unsigned int* brow0 = (const unsigned int*)&Bs[(waveN * 32 + l15) * ASTR];
        const unsigned int* brow1 = (const unsigned int*)&Bs[(waveN * 32 + 16 + l15) * ASTR];
        const int kb2 = half * 4;  // uint offset of kbase
#pragma unroll
        for (int p = 0; p < 4; ++p) {
            af.u[p]      = arow[kb2 + p];
            af.u[p + 4]  = arow[8 + kb2 + p];
            bf0.u[p]     = brow0[kb2 + p];
            bf0.u[p + 4] = brow0[8 + kb2 + p];
            bf1.u[p]     = brow1[kb2 + p];
            bf1.u[p + 4] = brow1[8 + kb2 + p];
        }
        acc0 = __builtin_amdgcn_wmma_f32_16x16x32_bf16(false, af.v, false, bf0.v,
                                                       (short)0, acc0, false, false);
        acc1 = __builtin_amdgcn_wmma_f32_16x16x32_bf16(false, af.v, false, bf1.v,
                                                       (short)0, acc1, false, false);
        __syncthreads();
    }

    // C/D layout: VGPR i -> M = i (lanes 0-15) / 8+i (lanes 16-31); N = lane&15
    const int rowOff = mBase + waveM * 16 + (half ? 8 : 0);
    const int col0   = nBase + waveN * 32 + l15;
#pragma unroll
    for (int i = 0; i < 8; ++i) {
        int r = rowOff + i;
        if (r < N) {
            float v0 = acc0[i], v1 = acc1[i];
            if (bias) { v0 += bias[col0]; v1 += bias[col0 + 16]; }
            if (relu) { v0 = fmaxf(v0, 0.0f); v1 = fmaxf(v1, 0.0f); }
            C[(long)r * F + col0]      = v0;
            C[(long)r * F + col0 + 16] = v1;
        }
    }
}

// ---------- edge scatter-add: agg[dst] += H[src] * norm_e ----------
__global__ void k_edge_scatter(const float* __restrict__ H, const int* __restrict__ src,
                               const int* __restrict__ dst, const float* __restrict__ ne,
                               float* __restrict__ agg, int E, int F) {
    long tid = (long)blockIdx.x * blockDim.x + threadIdx.x;
    int perEdge = F >> 2;                   // float4 chunks per edge
    long e = tid / perEdge;
    if (e >= E) return;
    int c = (int)(tid % perEdge) * 4;
    int s = src[e], d = dst[e];
    float w = ne[e];
    float4 v = *(const float4*)&H[(long)s * F + c];
    float* o = &agg[(long)d * F + c];
    atomicAdd(o + 0, v.x * w);
    atomicAdd(o + 1, v.y * w);
    atomicAdd(o + 2, v.z * w);
    atomicAdd(o + 3, v.w * w);
}

// out = agg + h*norm_self + bias (+relu), elementwise over N*F
__global__ void k_combine(const float* __restrict__ agg, const float* __restrict__ h,
                          const float* __restrict__ nself, const float* __restrict__ bias,
                          float* __restrict__ out, int N, int F, int relu) {
    long tid = (long)blockIdx.x * blockDim.x + threadIdx.x;
    if (tid >= (long)N * F) return;
    int i = (int)(tid / F), f = (int)(tid % F);
    float v = agg[tid] + h[tid] * nself[i] + bias[f];
    out[tid] = relu ? fmaxf(v, 0.0f) : v;
}

// segment max pool (values >= 0 post-relu; pooled pre-zeroed) via uint atomicMax
__global__ void k_pool_max(const float* __restrict__ h, const int* __restrict__ batch,
                           float* __restrict__ pooled, int N, int F) {
    long tid = (long)blockIdx.x * blockDim.x + threadIdx.x;
    if (tid >= (long)N * F) return;
    int i = (int)(tid / F), f = (int)(tid % F);
    atomicMax((unsigned int*)&pooled[(long)batch[i] * F + f], __float_as_uint(h[tid]));
}

// cat[N, 2F] = [loc | global_info[batch]]
__global__ void k_concat(const float* __restrict__ loc, const float* __restrict__ gi,
                         const int* __restrict__ batch, float* __restrict__ cat,
                         int N, int F) {
    long tid = (long)blockIdx.x * blockDim.x + threadIdx.x;
    long tot = (long)N * 2 * F;
    if (tid >= tot) return;
    int i = (int)(tid / (2 * F)), f = (int)(tid % (2 * F));
    cat[tid] = (f < F) ? loc[(long)i * F + f] : gi[(long)batch[i] * F + (f - F)];
}

// per-node dot with W3[HID] — one wave per node
__global__ void k_dot_w3(const float* __restrict__ h2, const float* __restrict__ W3,
                         float* __restrict__ out, int N) {
    int w = (blockIdx.x * blockDim.x + threadIdx.x) >> 5;
    int lane = threadIdx.x & 31;
    if (w >= N) return;
    float s = 0.0f;
#pragma unroll
    for (int f = lane; f < HID; f += 32) s += h2[(long)w * HID + f] * W3[f];
#pragma unroll
    for (int off = 16; off; off >>= 1) s += __shfl_down(s, off, 32);
    if (lane == 0) out[w] = s;
}

__global__ void k_edge_scatter1(const float* __restrict__ hv, const int* __restrict__ src,
                                const int* __restrict__ dst, const float* __restrict__ ne,
                                float* __restrict__ agg, int E) {
    int e = blockIdx.x * blockDim.x + threadIdx.x;
    if (e < E) atomicAdd(&agg[dst[e]], hv[src[e]] * ne[e]);
}

__global__ void k_combine1(const float* __restrict__ agg, const float* __restrict__ hv,
                           const float* __restrict__ nself, const float* __restrict__ b3,
                           float* __restrict__ out, int N) {
    int i = blockIdx.x * blockDim.x + threadIdx.x;
    if (i < N) out[i] = agg[i] + hv[i] * nself[i] + b3[0];
}

// ---------- driver ----------
extern "C" void kernel_launch(void* const* d_in, const int* in_sizes, int n_in,
                              void* d_out, int out_size, void* d_ws, size_t ws_size,
                              hipStream_t stream) {
    const float* x     = (const float*)d_in[0];
    const int*   src   = (const int*)  d_in[1];
    const int*   dst   = (const int*)  d_in[2];
    const int*   batch = (const int*)  d_in[3];
    const float* W0 = (const float*)d_in[4];  const float* b0 = (const float*)d_in[5];
    const float* Wf = (const float*)d_in[6];  const float* bf = (const float*)d_in[7];
    const float* W1 = (const float*)d_in[8];  const float* b1 = (const float*)d_in[9];
    const float* W2 = (const float*)d_in[10]; const float* b2 = (const float*)d_in[11];
    const float* W3 = (const float*)d_in[12]; const float* b3 = (const float*)d_in[13];
    float* out = (float*)d_out;

    // workspace carve-out
    char* wp = (char*)d_ws;
    auto carve = [&](size_t bytes) -> float* {
        float* p = (float*)wp;
        wp += (bytes + 255) & ~(size_t)255;
        return p;
    };
    float* dinv   = carve((size_t)NN * 4);
    float* nself  = carve((size_t)NN * 4);
    float* ne     = carve((size_t)NE * 4);
    float* bufA   = carve((size_t)NN * HID * 4);
    float* bufB   = carve((size_t)NN * HID * 4);
    float* bufAgg = carve((size_t)NN * HID * 4);
    float* bufCat = carve((size_t)NN * 2 * HID * 4);
    float* pooled = carve((size_t)NG * HID * 4);
    float* ginfo  = carve((size_t)NG * HID * 4);
    float* hv     = carve((size_t)NN * 4);
    float* agg1   = carve((size_t)NN * 4);

    const int T = 256;
    auto blks = [&](long n) { return (unsigned)((n + T - 1) / T); };
    const long NF  = (long)NN * HID;
    const long NF2 = (long)NN * 2 * HID;
    const long EW  = (long)NE * (HID / 4);
    dim3 gGemm(HID / 64, (NN + 63) / 64);   // 4 x 782
    dim3 gFc(HID / 64, 1);

    // norms
    k_zero<<<blks(NN), T, 0, stream>>>(dinv, NN);
    k_deg<<<blks(NE), T, 0, stream>>>(dst, dinv, NE);
    k_node_norms<<<blks(NN), T, 0, stream>>>(dinv, nself, NN);
    k_edge_norms<<<blks(NE), T, 0, stream>>>(src, dst, dinv, ne, NE);

    // conv0: h = relu(GCN(x, W0, b0))
    k_gemm_wmma<<<gGemm, T, 0, stream>>>(x, W0, nullptr, bufA, NN, FEAT, HID, 0);
    k_zero<<<blks(NF), T, 0, stream>>>(bufAgg, (int)NF);
    k_edge_scatter<<<blks(EW), T, 0, stream>>>(bufA, src, dst, ne, bufAgg, NE, HID);
    k_combine<<<blks(NF), T, 0, stream>>>(bufAgg, bufA, nself, b0, bufA, NN, HID, 1);

    // global max pool + fc1
    k_zero<<<blks((long)NG * HID), T, 0, stream>>>(pooled, NG * HID);
    k_pool_max<<<blks(NF), T, 0, stream>>>(bufA, batch, pooled, NN, HID);
    k_gemm_wmma<<<gFc, T, 0, stream>>>(pooled, Wf, bf, ginfo, NG, HID, HID, 0);

    // conv1 (1st): loc = relu(GCN(h, W1, b1))
    k_gemm_wmma<<<gGemm, T, 0, stream>>>(bufA, W1, nullptr, bufB, NN, HID, HID, 0);
    k_zero<<<blks(NF), T, 0, stream>>>(bufAgg, (int)NF);
    k_edge_scatter<<<blks(EW), T, 0, stream>>>(bufB, src, dst, ne, bufAgg, NE, HID);
    k_combine<<<blks(NF), T, 0, stream>>>(bufAgg, bufB, nself, b1, bufB, NN, HID, 1);

    // conv1 (2nd, shared weights)
    k_gemm_wmma<<<gGemm, T, 0, stream>>>(bufB, W1, nullptr, bufA, NN, HID, HID, 0);
    k_zero<<<blks(NF), T, 0, stream>>>(bufAgg, (int)NF);
    k_edge_scatter<<<blks(EW), T, 0, stream>>>(bufA, src, dst, ne, bufAgg, NE, HID);
    k_combine<<<blks(NF), T, 0, stream>>>(bufAgg, bufA, nself, b1, bufA, NN, HID, 1);

    // concat [loc | global_info[batch]]
    k_concat<<<blks(NF2), T, 0, stream>>>(bufA, ginfo, batch, bufCat, NN, HID);

    // conv2: h2 = relu(GCN(cat, W2, b2))
    k_gemm_wmma<<<gGemm, T, 0, stream>>>(bufCat, W2, nullptr, bufB, NN, 2 * HID, HID, 0);
    k_zero<<<blks(NF), T, 0, stream>>>(bufAgg, (int)NF);
    k_edge_scatter<<<blks(EW), T, 0, stream>>>(bufB, src, dst, ne, bufAgg, NE, HID);
    k_combine<<<blks(NF), T, 0, stream>>>(bufAgg, bufB, nself, b2, bufB, NN, HID, 1);

    // conv3: logits = GCN(h2, W3, b3) -> [N,1]
    k_dot_w3<<<blks((long)NN * 32), T, 0, stream>>>(bufB, W3, hv, NN);
    k_zero<<<blks(NN), T, 0, stream>>>(agg1, NN);
    k_edge_scatter1<<<blks(NE), T, 0, stream>>>(hv, src, dst, ne, agg1, NE);
    k_combine1<<<blks(NN), T, 0, stream>>>(agg1, hv, nself, b3, out, NN);
}
